// DSConv_59442347377175
// MI455X (gfx1250) — matrix operand order repaired and checked by reading, not compiled, not run on hardware
//
#include <hip/hip_runtime.h>
#include <hip/hip_bf16.h>

typedef __attribute__((ext_vector_type(16))) _Float16 v16h;
typedef __attribute__((ext_vector_type(8)))  _Float16 v8h;
typedef __attribute__((ext_vector_type(8)))  float    v8f;

// Problem constants
#define BN 2
#define CC 64
#define HH 256
#define WW 256
#define KK 9

// ---- workspace layout (bytes) ----
#define XH_OFF  ((size_t)0)                                   // x as NHWC f16: B*H*W*64
#define XH_BYTES ((size_t)BN*HH*WW*CC*2)                      // 16 MB
#define WOH_OFF (XH_OFF + XH_BYTES)                           // w_off as [oc][tap][c] f16
#define WOH_BYTES ((size_t)18*9*64*2)                         // 20736
#define WKH_OFF (WOH_OFF + WOH_BYTES)                         // w_dsc as [k][co][c] f16
#define WKH_BYTES ((size_t)9*64*64*2)                         // 73728
#define OFF_OFF (WKH_OFF + WKH_BYTES)                         // offset conv out f32 B*18*H*W
#define OFF_BYTES ((size_t)BN*18*HH*WW*4)
#define YC_OFF  (OFF_OFF + OFF_BYTES)                         // ycoord f32 B*9*H*W
#define YC_BYTES ((size_t)BN*9*HH*WW*4)
#define YB_OFF  (YC_OFF + YC_BYTES)                           // pre-GN conv out f32 B*64*H*W
#define YB_BYTES ((size_t)BN*CC*HH*WW*4)
#define ST_OFF  (YB_OFF + YB_BYTES)                           // stats floats (200)
// stats float indices: [0,18) gnoSum  [18,36) gnoSq  [36,54) gnoMean [54,72) gnoRstd
//                      [72,104) gSum [104,136) gSq [136,168) gMean [168,200) gRstd

// Async global->LDS staging (CDNA5 GLOBAL_LOAD_ASYNC_TO_LDS_B128, ASYNCcnt-tracked)
#define USE_ASYNC_LDS 1

__device__ __forceinline__ void async_b128(void* lds, const void* gmem) {
    unsigned la = (unsigned)(size_t)lds;        // low 32 bits of generic = LDS offset
    unsigned long long ga = (unsigned long long)gmem;
    asm volatile("global_load_async_to_lds_b128 %0, %1, off"
                 :: "v"(la), "v"(ga) : "memory");
}
__device__ __forceinline__ void async_wait0() {
    asm volatile("s_wait_asynccnt 0x0" ::: "memory");
}

// Fragment load helpers (layouts per CDNA5 ISA):
//  A 16x32 f16: lanes 0-15 (M=lane): K = kb+{0..7} and kb+16+{0..7}; lanes 16-31 shift K by 8
__device__ __forceinline__ v16h load_fragA(const _Float16* row, int kb, int half) {
    v8h lo = *(const v8h*)(row + kb + half * 8);
    v8h hi = *(const v8h*)(row + kb + 16 + half * 8);
    return __builtin_shufflevector(lo, hi, 0,1,2,3,4,5,6,7,8,9,10,11,12,13,14,15);
}
//  B 32x16 f16: same K order as A: K = kb+{0..7} and kb+16+{0..7}; lanes 16-31 shift K by 8
__device__ __forceinline__ v16h load_fragB(const _Float16* row, int kb, int half) {
    v8h lo = *(const v8h*)(row + kb + half * 8);
    v8h hi = *(const v8h*)(row + kb + 16 + half * 8);
    return __builtin_shufflevector(lo, hi, 0,1,2,3,4,5,6,7,8,9,10,11,12,13,14,15);
}

// ---------------- zero the atomic accumulators (every launch) ----------------
__global__ void kz_zero(float* st) {
    int t = threadIdx.x;
    if (t < 18) { st[t] = 0.f; st[18 + t] = 0.f; }
    if (t < 32) { st[72 + t] = 0.f; st[104 + t] = 0.f; }
}

// ---------------- P1: x NCHW f32 -> NHWC f16 ----------------
__global__ __launch_bounds__(256) void kp1_xh(const float* __restrict__ x,
                                              _Float16* __restrict__ xh) {
    int idx = blockIdx.x * 256 + threadIdx.x;      // b*65536 + h*256 + w
    int pix = idx & 65535;
    int b   = idx >> 16;
    size_t dst = (size_t)idx * 64;
    #pragma unroll 8
    for (int c = 0; c < 64; ++c)
        xh[dst + c] = (_Float16)x[((size_t)(b * 64 + c) << 16) + pix];
}

// ---------------- P2: w_dsc (co,c,k,1) f32 -> (k,co,c) f16 ----------------
__global__ void kp2_wk(const float* __restrict__ w_dsc, _Float16* __restrict__ wkh) {
    int idx = blockIdx.x * 256 + threadIdx.x;      // (k*64+co)*64 + c
    if (idx >= 9 * 64 * 64) return;
    int c  = idx & 63;
    int co = (idx >> 6) & 63;
    int k  = idx >> 12;
    wkh[idx] = (_Float16)w_dsc[((size_t)(co * 64 + c)) * 9 + k];
}

// ---------------- P3: w_off (oc,c,3,3) f32 -> (oc,tap,c) f16 ----------------
__global__ void kp3_wo(const float* __restrict__ w_off, _Float16* __restrict__ woh) {
    int idx = blockIdx.x * 256 + threadIdx.x;      // (oc*9+t)*64 + c
    if (idx >= 18 * 9 * 64) return;
    int c  = idx & 63;
    int t  = (idx >> 6) % 9;
    int oc = idx / 576;
    woh[idx] = (_Float16)w_off[((size_t)(oc * 64 + c)) * 9 + t];
}

// ---------------- K1: 3x3 offset conv (64 -> 18) via WMMA ----------------
// Block = 256 threads (8 waves). Tile: M=32 (18 used), N=64 pixels, K=576 (9 taps x 64 ch).
__global__ __launch_bounds__(256) void k1_offconv(const _Float16* __restrict__ xh,
                                                  const _Float16* __restrict__ woh,
                                                  const float* __restrict__ b_off,
                                                  float* __restrict__ offs) {
    __shared__ __attribute__((aligned(64))) _Float16 sA[32 * 64];  // [oc][c] (pad oc>=18 = 0)
    __shared__ __attribute__((aligned(64))) _Float16 sB[64 * 64];  // [n][c]

    int t  = threadIdx.x;
    int bx = blockIdx.x;                    // B*H*(W/64)
    int wt = bx & 3;
    int h  = (bx >> 2) & 255;
    int b  = bx >> 10;
    int w0 = wt * 64;

    int lane = t & 31, wv = t >> 5;
    int mt = wv & 1, nt = wv >> 1;          // M tile (2), N tile (4)
    int half = lane >> 4, r = lane & 15;

    v8f acc = {};

    for (int tap = 0; tap < 9; ++tap) {
        int ty = tap / 3, tx = tap % 3;
        // stage A: 32x64 f16, one v8h per thread (zeros for oc >= 18)
        {
            int oc = t >> 3, cc = t & 7;
            v8h av = {};
            if (oc < 18) av = *(const v8h*)&woh[((size_t)(oc * 9 + tap) << 6) + (cc << 3)];
            *(v8h*)&sA[(oc << 6) + (cc << 3)] = av;
        }
        // stage B: 64 px x 64 ch, two v8h per thread (zero pad at borders)
        int y = h + ty - 1;
        #pragma unroll
        for (int j = 0; j < 2; ++j) {
            int task = t + 256 * j;
            int n = task >> 3, cc = task & 7;
            int xx = w0 + n + tx - 1;
            v8h bv = {};
            if ((unsigned)y < 256u && (unsigned)xx < 256u)
                bv = *(const v8h*)&xh[(((size_t)(b * 256 + y) * 256 + xx) << 6) + (cc << 3)];
            *(v8h*)&sB[(n << 6) + (cc << 3)] = bv;
        }
        __syncthreads();
        #pragma unroll
        for (int kb = 0; kb < 64; kb += 32) {
            v16h A  = load_fragA(&sA[(16 * mt + r) << 6], kb, half);
            v16h Bm = load_fragB(&sB[(16 * nt + r) << 6], kb, half);
            acc = __builtin_amdgcn_wmma_f32_16x16x32_f16(
                      false, A, false, Bm, (short)0, acc, false, false);
        }
        __syncthreads();
    }

    int nn = w0 + 16 * nt + r;
    #pragma unroll
    for (int i = 0; i < 8; ++i) {
        int co = 16 * mt + i + 8 * half;
        if (co < 18)
            offs[((size_t)(b * 18 + co) << 16) + (h << 8) + nn] = acc[i] + b_off[co];
    }
}

// ---------------- K1b: GN stats for offset (groups=9, 2 ch each) ----------------
__global__ __launch_bounds__(256) void k1b_stats(const float* __restrict__ offs,
                                                 float* __restrict__ st) {
    int t  = threadIdx.x;
    int bx = blockIdx.x;                   // B*9*16 blocks
    int slice = bx & 15;
    int g = (bx >> 4) % 9;
    int b = bx / 144;
    float s = 0.f, q = 0.f;
    for (int i = 0; i < 32; ++i) {
        int e   = slice * 8192 + i * 256 + t;       // 0..131071 within group
        int ch  = 2 * g + (e >> 16);
        int pos = e & 65535;
        float v = offs[((size_t)(b * 18 + ch) << 16) + pos];
        s += v; q += v * v;
    }
    __shared__ float rs[256], rq[256];
    rs[t] = s; rq[t] = q;
    __syncthreads();
    for (int st2 = 128; st2 > 0; st2 >>= 1) {
        if (t < st2) { rs[t] += rs[t + st2]; rq[t] += rq[t + st2]; }
        __syncthreads();
    }
    if (t == 0) {
        atomicAdd(&st[b * 9 + g], rs[0]);
        atomicAdd(&st[18 + b * 9 + g], rq[0]);
    }
}

// ---------------- K1c: finalize offset GN stats ----------------
__global__ void k1c_fin(float* st) {
    int t = threadIdx.x;
    if (t < 18) {
        const float N = 131072.f;
        float m = st[t] / N;
        float v = st[18 + t] / N - m * m;
        st[36 + t] = m;
        st[54 + t] = rsqrtf(v + 1e-5f);
    }
}

// ---------------- K2: GN + tanh + outward cumsum -> clipped y coordinates ----------------
__global__ __launch_bounds__(256) void k2_coords(const float* __restrict__ offs,
                                                 const float* __restrict__ gno_w,
                                                 const float* __restrict__ gno_b,
                                                 const float* __restrict__ st,
                                                 float* __restrict__ ycoord) {
    int idx = blockIdx.x * 256 + threadIdx.x;      // b*65536 + pix
    int pix = idx & 65535;
    int b   = idx >> 16;
    float v[9];
    #pragma unroll
    for (int k = 0; k < 9; ++k) {
        int g = k >> 1;
        float m  = st[36 + b * 9 + g];
        float rs = st[54 + b * 9 + g];
        float o  = offs[((size_t)(b * 18 + k) << 16) + pix];
        v[k] = tanhf((o - m) * rs * gno_w[k] + gno_b[k]);
    }
    float hc = (float)(pix >> 8);
    float cum[9];
    cum[4] = 0.f;
    float s = 0.f;
    #pragma unroll
    for (int k = 3; k >= 0; --k) { s += v[k]; cum[k] = s; }
    s = 0.f;
    #pragma unroll
    for (int k = 5; k < 9; ++k) { s += v[k]; cum[k] = s; }
    #pragma unroll
    for (int k = 0; k < 9; ++k) {
        float y = hc + cum[k];
        y = fminf(fmaxf(y, 0.f), 255.f);
        ycoord[((size_t)(b * 9 + k) << 16) + pix] = y;
    }
}

// ---------------- K3: fused gather + per-tap 64x64 GEMM via WMMA f16 ----------------
// Block = 256 threads (8 waves). Tile: M=64 (C_OUT), N=64 pixels, K=576 (9 taps x 64 ch).
// Per tap k: W_k async-DMAs into LDS while gather (y-lerp) stages s_k[n][c];
// each wave: 1 A frag reused over 2 B tiles -> 4 WMMA per tap.
__global__ __launch_bounds__(256) void k3_conv(const _Float16* __restrict__ xh,
                                               const _Float16* __restrict__ wkh,
                                               const float* __restrict__ ycoord,
                                               const float* __restrict__ b_dsc,
                                               float* __restrict__ ybuf,
                                               float* __restrict__ st) {
    __shared__ __attribute__((aligned(64))) _Float16 sW[64 * 64];  // [co][c]
    __shared__ __attribute__((aligned(64))) _Float16 sS[64 * 64];  // [n][c]
    __shared__ float sSum[16], sSq[16];

    int t  = threadIdx.x;
    int bx = blockIdx.x;                    // B*H*(W/64)
    int wt = bx & 3;
    int h  = (bx >> 2) & 255;
    int b  = bx >> 10;
    int w0 = wt * 64;

    if (t < 16) { sSum[t] = 0.f; sSq[t] = 0.f; }

    int lane = t & 31, wv = t >> 5;
    int mt = wv & 3;                        // M tile (4)
    int nt0 = (wv >> 2) * 2;                // N tiles {nt0, nt0+1} of 4
    int half = lane >> 4, r = lane & 15;

    v8f acc0 = {}, acc1 = {};

    for (int k = 0; k < 9; ++k) {
        // stage W_k (64x64 f16 = 8KB): 512 x 16B chunks, 2 per thread
#if USE_ASYNC_LDS
        async_b128(&sW[t * 8],         wkh + (size_t)k * 4096 + t * 8);
        async_b128(&sW[(t + 256) * 8], wkh + (size_t)k * 4096 + (t + 256) * 8);
#else
        {
            const uint4* wg = (const uint4*)(wkh + (size_t)k * 4096);
            uint4* sw = (uint4*)sW;
            sw[t]       = wg[t];
            sw[t + 256] = wg[t + 256];
        }
#endif
        // gather: 64 px x 64 ch; two 128B NHWC rows per sample, lerp in y
        #pragma unroll
        for (int j = 0; j < 2; ++j) {
            int task = t + 256 * j;
            int gp = task >> 3, cc = task & 7;
            int xk = w0 + gp + k - 4;
            xk = xk < 0 ? 0 : (xk > 255 ? 255 : xk);
            float y = ycoord[(((size_t)(b * 9 + k) * 256 + h) << 8) + w0 + gp];
            int y0 = (int)y;
            float wy1 = y - (float)y0;
            int y1 = y0 + 1; if (y1 > 255) y1 = 255;
            const _Float16* p0 = xh + (((size_t)(b * 256 + y0) * 256 + xk) << 6) + (cc << 3);
            const _Float16* p1 = xh + (((size_t)(b * 256 + y1) * 256 + xk) << 6) + (cc << 3);
            v8h v0 = *(const v8h*)p0;
            v8h v1 = *(const v8h*)p1;
            float wy0 = 1.f - wy1;
            v8h res;
            #pragma unroll
            for (int i = 0; i < 8; ++i)
                res[i] = (_Float16)(wy0 * (float)v0[i] + wy1 * (float)v1[i]);
            *(v8h*)&sS[(gp << 6) + (cc << 3)] = res;
        }
#if USE_ASYNC_LDS
        async_wait0();
#endif
        __syncthreads();

        #pragma unroll
        for (int kb = 0; kb < 64; kb += 32) {
            v16h A  = load_fragA(&sW[(16 * mt + r) << 6], kb, half);
            v16h B0 = load_fragB(&sS[(16 * nt0 + r) << 6], kb, half);
            v16h B1 = load_fragB(&sS[(16 * (nt0 + 1) + r) << 6], kb, half);
            acc0 = __builtin_amdgcn_wmma_f32_16x16x32_f16(
                       false, A, false, B0, (short)0, acc0, false, false);
            acc1 = __builtin_amdgcn_wmma_f32_16x16x32_f16(
                       false, A, false, B1, (short)0, acc1, false, false);
        }
        __syncthreads();
    }

    // C/D layout: VGPR i, lanes0-15 -> M=i, lanes16-31 -> M=i+8; N = lane%16
    #pragma unroll
    for (int j = 0; j < 2; ++j) {
        v8f acc = j ? acc1 : acc0;
        int nn = w0 + 16 * (nt0 + j) + r;
        #pragma unroll
        for (int i = 0; i < 8; ++i) {
            int co = 16 * mt + i + 8 * half;
            float val = acc[i] + b_dsc[co];
            ybuf[((size_t)(b * 64 + co) << 16) + (h << 8) + nn] = val;
            atomicAdd(&sSum[co >> 2], val);
            atomicAdd(&sSq[co >> 2], val * val);
        }
    }
    __syncthreads();
    if (t < 16) {
        atomicAdd(&st[72 + b * 16 + t], sSum[t]);
        atomicAdd(&st[104 + b * 16 + t], sSq[t]);
    }
}

// ---------------- K4: finalize output GN stats ----------------
__global__ void k4_fin(float* st) {
    int t = threadIdx.x;
    if (t < 32) {
        const float N = 262144.f;           // 4 ch * 65536
        float m = st[72 + t] / N;
        float v = st[104 + t] / N - m * m;
        st[136 + t] = m;
        st[168 + t] = rsqrtf(v + 1e-5f);
    }
}

// ---------------- K5: apply GN(groups=16) + ReLU ----------------
__global__ __launch_bounds__(256) void k5_gnrelu(const float* __restrict__ ybuf,
                                                 const float* __restrict__ gn_w,
                                                 const float* __restrict__ gn_b,
                                                 const float* __restrict__ st,
                                                 float* __restrict__ out) {
    int idx = blockIdx.x * 256 + threadIdx.x;      // b*64*65536 + c*65536 + pix
    int c = (idx >> 16) & 63;
    int b = idx >> 22;
    int g = c >> 2;
    float m  = st[136 + b * 16 + g];
    float rs = st[168 + b * 16 + g];
    float v = (ybuf[idx] - m) * rs * gn_w[c] + gn_b[c];
    out[idx] = fmaxf(v, 0.f);
}

extern "C" void kernel_launch(void* const* d_in, const int* in_sizes, int n_in,
                              void* d_out, int out_size, void* d_ws, size_t ws_size,
                              hipStream_t stream) {
    const float* x     = (const float*)d_in[0];
    const float* w_off = (const float*)d_in[1];
    const float* b_off = (const float*)d_in[2];
    const float* gno_w = (const float*)d_in[3];
    const float* gno_b = (const float*)d_in[4];
    const float* w_dsc = (const float*)d_in[5];
    const float* b_dsc = (const float*)d_in[6];
    const float* gn_w  = (const float*)d_in[7];
    const float* gn_b  = (const float*)d_in[8];
    float* out = (float*)d_out;

    char* ws = (char*)d_ws;
    _Float16* xh   = (_Float16*)(ws + XH_OFF);
    _Float16* woh  = (_Float16*)(ws + WOH_OFF);
    _Float16* wkh  = (_Float16*)(ws + WKH_OFF);
    float*    offs = (float*)(ws + OFF_OFF);
    float*    yc   = (float*)(ws + YC_OFF);
    float*    ybuf = (float*)(ws + YB_OFF);
    float*    st   = (float*)(ws + ST_OFF);

    kz_zero<<<1, 128, 0, stream>>>(st);
    kp1_xh<<<(BN * HH * WW) / 256, 256, 0, stream>>>(x, xh);
    kp2_wk<<<(9 * 64 * 64 + 255) / 256, 256, 0, stream>>>(w_dsc, wkh);
    kp3_wo<<<(18 * 9 * 64 + 255) / 256, 256, 0, stream>>>(w_off, woh);
    k1_offconv<<<BN * HH * (WW / 64), 256, 0, stream>>>(xh, woh, b_off, offs);
    k1b_stats<<<BN * 9 * 16, 256, 0, stream>>>(offs, st);
    k1c_fin<<<1, 32, 0, stream>>>(st);
    k2_coords<<<(BN * HH * WW) / 256, 256, 0, stream>>>(offs, gno_w, gno_b, st, yc);
    k3_conv<<<BN * HH * (WW / 64), 256, 0, stream>>>(xh, wkh, yc, b_dsc, ybuf, st);
    k4_fin<<<1, 32, 0, stream>>>(st);
    k5_gnrelu<<<(BN * CC * HH * WW) / 256, 256, 0, stream>>>(ybuf, gn_w, gn_b, st, out);
    (void)in_sizes; (void)n_in; (void)out_size; (void)ws_size;
}
